// TransNFCM_25632364822733
// MI455X (gfx1250) — matrix operand order repaired and checked by reading, not compiled, not run on hardware
//
#include <hip/hip_runtime.h>

// ---------------------------------------------------------------------------
// SoftTriple-style loss, MI455X (gfx1250, wave32).
// Hot path: 2048 x 32768 x 512 GEMM (68.7 GFLOP) -> bf16 WMMA 16x16x32,
// fused with the K=4 softmax-similarity so logits (256 MB) never exist.
// Data staged global->LDS with async DMA (global_load_async_to_lds_b128,
// ASYNCcnt) into double-buffered LDS; B operand built via plain ds_load_b128.
// Whole working set is L2-resident (192 MB L2).
// ---------------------------------------------------------------------------

typedef __attribute__((ext_vector_type(16))) __bf16 v16bf;
typedef __attribute__((ext_vector_type(8)))  __bf16 v8bf;
typedef __attribute__((ext_vector_type(8)))  float  v8f;

#define B_ROWS    2048
#define DIM       512
#define CATS      8192
#define CK        32768       // CATS * 4
#define INV_GAMMA 10.0f       // 1 / 0.1
#define LAM       0.1f
#define MARGIN    0.01f
#define TAU       0.2f

__device__ __forceinline__ unsigned bfbits(float f) {
  union { float f; unsigned u; } a; a.f = f;
  return (a.u + 0x7FFFu + ((a.u >> 16) & 1u)) >> 16;         // RNE
}
__device__ __forceinline__ unsigned pk2(float lo, float hi) {
  return bfbits(lo) | (bfbits(hi) << 16);
}

// softmax_k(l/gamma) . l over a 4-lane group (4 sub-centers sit in 4 adjacent
// lanes of the WMMA C layout). Result replicated across the group.
__device__ __forceinline__ float softsim4(float l) {
  float m = l;
  m = fmaxf(m, __shfl_xor(m, 1, 32));
  m = fmaxf(m, __shfl_xor(m, 2, 32));
  float e  = __expf((l - m) * INV_GAMMA);
  float s1 = e, s2 = e * l;
  s1 += __shfl_xor(s1, 1, 32);  s2 += __shfl_xor(s2, 1, 32);
  s1 += __shfl_xor(s1, 2, 32);  s2 += __shfl_xor(s2, 2, 32);
  return s2 / s1;
}

// ---------------------------------------------------------------------------
__global__ void zero_kernel(float* __restrict__ tot, float* __restrict__ accums) {
  int t = blockIdx.x * 256 + threadIdx.x;
  if (t < B_ROWS) tot[t] = 0.0f;
  if (blockIdx.x == 0 && threadIdx.x < 2) accums[threadIdx.x] = 0.0f;
}

// f32 -> bf16 one-shot conversion (packed 64-bit stores). n multiple of 8.
__global__ __launch_bounds__(256) void cvt_kernel(const float* __restrict__ src,
                                                  __bf16* __restrict__ dst, int n) {
  int i = (blockIdx.x * 256 + threadIdx.x) * 8;
  if (i >= n) return;
  float4 a = *(const float4*)&src[i];
  float4 b = *(const float4*)&src[i + 4];
  uint2 p0; p0.x = pk2(a.x, a.y); p0.y = pk2(a.z, a.w);
  uint2 p1; p1.x = pk2(b.x, b.y); p1.y = pk2(b.z, b.w);
  *(uint2*)&dst[i]     = p0;
  *(uint2*)&dst[i + 4] = p1;
}

// ---------------------------------------------------------------------------
// Per-category rowsum inverse + sqrt(2-2<wn_k,wn_{k+1}>) regularizer (f32 W).
__global__ __launch_bounds__(256) void wstats_kernel(
    const float* __restrict__ W, float* __restrict__ rsinv,
    float* __restrict__ accums) {
  const int lane = threadIdx.x & 31;
  const int wid  = threadIdx.x >> 5;
  const int c    = blockIdx.x * 8 + wid;
  const float* w0 = W + (size_t)c * 4 * DIM;

  float s0 = 0, s1 = 0, s2 = 0, s3 = 0, d01 = 0, d12 = 0, d23 = 0;
  for (int d = lane; d < DIM; d += 32) {
    float a = w0[d], b = w0[DIM + d], cc = w0[2 * DIM + d], e = w0[3 * DIM + d];
    s0 += a; s1 += b; s2 += cc; s3 += e;
    d01 += a * b; d12 += b * cc; d23 += cc * e;
  }
#pragma unroll
  for (int m = 1; m < 32; m <<= 1) {
    s0  += __shfl_xor(s0,  m, 32);  s1  += __shfl_xor(s1,  m, 32);
    s2  += __shfl_xor(s2,  m, 32);  s3  += __shfl_xor(s3,  m, 32);
    d01 += __shfl_xor(d01, m, 32);  d12 += __shfl_xor(d12, m, 32);
    d23 += __shfl_xor(d23, m, 32);
  }
  if (lane == 0) {
    rsinv[c * 4 + 0] = 1.0f / s0;  rsinv[c * 4 + 1] = 1.0f / s1;
    rsinv[c * 4 + 2] = 1.0f / s2;  rsinv[c * 4 + 3] = 1.0f / s3;
    float r = sqrtf(fmaxf(0.0f, 2.0f - 2.0f * d01 / (s0 * s1)))
            + sqrtf(fmaxf(0.0f, 2.0f - 2.0f * d12 / (s1 * s2)))
            + sqrtf(fmaxf(0.0f, 2.0f - 2.0f * d23 / (s2 * s3)));
    atomicAdd(&accums[1], r);
  }
}

// ---------------------------------------------------------------------------
// Fused GEMM: 64(M) x 128(N) block tile, K-chunks of 32, 8 waves (wave32).
// BF16SRC: async global->LDS DMA, double-buffered; else f32 + in-flight cvt.
// B operand via plain ds_load_b128 from row-major LDS.
// LOSS=0: accumulate tot[b]=sum_c S_norm.  LOSS=1: gathered columns
// cat[j]*4+k, full loss term -log1p(base/(base+expt)).
#define LDT 40    // padded bf16 row stride (80 B: 16B-aligned, conflict-free)

template <bool BF16SRC, bool LOSS>
__global__ __launch_bounds__(256) void gemm_kernel(
    const void* __restrict__ xsrc, const void* __restrict__ wsrc,
    const int* __restrict__ cat, const float* __restrict__ rsinv,
    float* __restrict__ tot, float* __restrict__ accums) {
  __shared__ __align__(16) __bf16 sA[2][64 * LDT];
  __shared__ __align__(16) __bf16 sB[2][128 * LDT];
  __shared__ int   srow[128];
  __shared__ float wsum[8];

  const int tid  = threadIdx.x;
  const int lane = tid & 31, wid = tid >> 5;
  const int wm = wid >> 1, wn = wid & 1;
  const int mbase = blockIdx.y * 64;
  const int nbase = blockIdx.x * 128;            // CK base (LOSS=0)
  const int jbase = blockIdx.x * 32;             // cat base (LOSS=1)

  if constexpr (LOSS) {
    if (tid < 128) srow[tid] = cat[jbase + (tid >> 2)] * 4 + (tid & 3);
    __syncthreads();
  }
  auto brow = [&](int r) -> int {
    if constexpr (LOSS) return srow[r]; else return nbase + r;
  };

  v8f acc[4];
#pragma unroll
  for (int t = 0; t < 4; ++t) acc[t] = (v8f){0, 0, 0, 0, 0, 0, 0, 0};

  // WMMA consumption of one LDS tile pair.
  auto computeStep = [&](int buf) {
    const int mrow = wm * 16 + (lane & 15);
    const int koff = (lane < 16) ? 0 : 8;
    v16bf af;
#pragma unroll
    for (int j = 0; j < 8; ++j) {
      af[j]     = sA[buf][mrow * LDT + koff + j];
      af[8 + j] = sA[buf][mrow * LDT + 16 + koff + j];
    }
#pragma unroll
    for (int t = 0; t < 4; ++t) {
      // B operand via plain LDS load: row-major [n][k] tile, lane n reads k run.
      unsigned ba = (unsigned)(size_t)&sB[buf][(wn * 64 + t * 16 + (lane & 15)) * LDT]
                    + (unsigned)((lane >> 4) * 16);
      v8bf blo, bhi;
      asm volatile("ds_load_b128 %0, %1"           : "=v"(blo) : "v"(ba));
      asm volatile("ds_load_b128 %0, %1 offset:32" : "=v"(bhi) : "v"(ba));
      asm volatile("s_wait_dscnt 0x0" : "+v"(blo), "+v"(bhi));   // order wait->wmma
      v16bf bfr = __builtin_shufflevector(blo, bhi,
                    0, 1, 2, 3, 4, 5, 6, 7, 8, 9, 10, 11, 12, 13, 14, 15);
      acc[t] = __builtin_amdgcn_wmma_f32_16x16x32_bf16(
          false, af, false, bfr, (short)0, acc[t], false, false);
    }
  };

  if constexpr (BF16SRC) {
    // ---- async DMA pipeline: 3 async b128 per thread per tile -------------
    const __bf16* xb = (const __bf16*)xsrc;
    const __bf16* wb = (const __bf16*)wsrc;
    auto issueAsync = [&](int buf, int dbase) {
      {
        int r = tid >> 2, q = tid & 3;
        unsigned           ld = (unsigned)(size_t)&sA[buf][r * LDT + q * 8];
        unsigned long long ga = (unsigned long long)(size_t)
                                &xb[(size_t)(mbase + r) * DIM + dbase + q * 8];
        asm volatile("global_load_async_to_lds_b128 %0, %1, off"
                     :: "v"(ld), "v"(ga) : "memory");
      }
#pragma unroll
      for (int i = 0; i < 2; ++i) {
        int idx = tid * 2 + i;
        int r = idx >> 2, q = idx & 3;
        unsigned           ld = (unsigned)(size_t)&sB[buf][r * LDT + q * 8];
        unsigned long long ga = (unsigned long long)(size_t)
                                &wb[(size_t)brow(r) * DIM + dbase + q * 8];
        asm volatile("global_load_async_to_lds_b128 %0, %1, off"
                     :: "v"(ld), "v"(ga) : "memory");
      }
    };

    issueAsync(0, 0);
#pragma unroll 1
    for (int dk = 0; dk < DIM / 32; ++dk) {
      __syncthreads();                         // buf[(dk+1)&1] free for reuse
      if (dk + 1 < DIM / 32) {
        issueAsync((dk + 1) & 1, (dk + 1) * 32);         // overlaps compute dk
        asm volatile("s_wait_asynccnt 0x3" ::: "memory"); // tile dk resident
      } else {
        asm volatile("s_wait_asynccnt 0x0" ::: "memory");
      }
      __syncthreads();                         // all waves' tile dk visible
      computeStep(dk & 1);
    }
  } else {
    // ---- f32 fallback: register pipeline + in-flight bf16 conversion ------
    const float* xf = (const float*)xsrc;
    const float* wf = (const float*)wsrc;
    float4 ra_f[2], rb_f[4];
    auto loadTile = [&](int dbase) {
#pragma unroll
      for (int i = 0; i < 2; ++i) {
        int idx = tid * 2 + i;
        ra_f[i] = *(const float4*)&xf[(size_t)(mbase + (idx >> 3)) * DIM + dbase + (idx & 7) * 4];
      }
#pragma unroll
      for (int i = 0; i < 4; ++i) {
        int idx = tid * 4 + i;
        rb_f[i] = *(const float4*)&wf[(size_t)brow(idx >> 3) * DIM + dbase + (idx & 7) * 4];
      }
      if (dbase + 64 < DIM)
        __builtin_prefetch(&wf[(size_t)brow(tid >> 1) * DIM + dbase + 64], 0, 1);
    };
    auto storeTile = [&]() {
#pragma unroll
      for (int i = 0; i < 2; ++i) {
        int idx = tid * 2 + i;
        uint2 p; p.x = pk2(ra_f[i].x, ra_f[i].y); p.y = pk2(ra_f[i].z, ra_f[i].w);
        *(uint2*)&sA[0][(idx >> 3) * LDT + (idx & 7) * 4] = p;
      }
#pragma unroll
      for (int i = 0; i < 4; ++i) {
        int idx = tid * 4 + i;
        uint2 p; p.x = pk2(rb_f[i].x, rb_f[i].y); p.y = pk2(rb_f[i].z, rb_f[i].w);
        *(uint2*)&sB[0][(idx >> 3) * LDT + (idx & 7) * 4] = p;
      }
    };

    loadTile(0);
#pragma unroll 1
    for (int dk = 0; dk < DIM / 32; ++dk) {
      __syncthreads();
      storeTile();
      __syncthreads();
      if (dk + 1 < DIM / 32) loadTile((dk + 1) * 32);
      computeStep(0);
    }
  }

  // ---- epilogues ---------------------------------------------------------
  if constexpr (!LOSS) {
    float rp[8];
#pragma unroll
    for (int p = 0; p < 8; ++p) rp[p] = 0.0f;
#pragma unroll
    for (int t = 0; t < 4; ++t) {
      const float inv = rsinv[nbase + wn * 64 + t * 16 + (lane & 15)];
#pragma unroll
      for (int p = 0; p < 8; ++p) {
        float S = softsim4(acc[t][p] * inv);
        float q = ((lane & 3) == 0) ? S : 0.0f;       // one per category group
        q += __shfl_xor(q, 1, 32);
        q += __shfl_xor(q, 2, 32);
        q += __shfl_xor(q, 4, 32);
        q += __shfl_xor(q, 8, 32);                    // sum of 4 categories/half
        rp[p] += q;
      }
    }
    if (lane == 0 || lane == 16) {
      const int rbase = mbase + wm * 16 + ((lane == 16) ? 8 : 0);
#pragma unroll
      for (int p = 0; p < 8; ++p) atomicAdd(&tot[rbase + p], rp[p]);
    }
  } else {
    const int rbase = mbase + wm * 16 + ((lane < 16) ? 0 : 8);
    float totv[8];
#pragma unroll
    for (int p = 0; p < 8; ++p) totv[p] = tot[rbase + p];

    float vsum = 0.0f;
#pragma unroll
    for (int t = 0; t < 4; ++t) {
      const float inv = rsinv[srow[wn * 64 + t * 16 + (lane & 15)]];
#pragma unroll
      for (int p = 0; p < 8; ++p) {
        float lr     = acc[t][p];
        float S_raw  = softsim4(lr);
        float S_norm = softsim4(lr * inv);
        float base   = __expf(LAM * (S_raw - MARGIN));
        float expt   = LAM * (totv[p] - S_norm);
        float val    = -__logf(1.0f + base / (base + expt));
        vsum += ((lane & 3) == 0) ? val : 0.0f;       // one per (row, j)
      }
    }
#pragma unroll
    for (int m = 1; m < 32; m <<= 1) vsum += __shfl_xor(vsum, m, 32);
    if (lane == 0) wsum[wid] = vsum;
    __syncthreads();
    if (tid == 0) {
      float s = 0.0f;
#pragma unroll
      for (int w = 0; w < 8; ++w) s += wsum[w];
      atomicAdd(&accums[0], s);
    }
  }
}

// ---------------------------------------------------------------------------
__global__ void finalize_kernel(const float* __restrict__ accums,
                                float* __restrict__ out) {
  float st  = accums[0] / ((float)B_ROWS * (float)B_ROWS);
  float reg = TAU * accums[1] / ((float)CATS * 4.0f * 3.0f);
  out[0] = st + reg;
}

// ---------------------------------------------------------------------------
// ws layout (floats): [0,2048) tot | [2048,34816) rsinv | [34816,34832) accums
// then 256B-aligned bf16 mirrors of x (2 MB) and W (32 MB) if ws permits.
#define XBF_OFF_F 34944u
#define WBF_OFF_F (XBF_OFF_F + (B_ROWS * DIM) / 2)
#define WS_NEED_BYTES ((size_t)(WBF_OFF_F + (CK * DIM) / 2) * 4u)

extern "C" void kernel_launch(void* const* d_in, const int* in_sizes, int n_in,
                              void* d_out, int out_size, void* d_ws, size_t ws_size,
                              hipStream_t stream) {
  (void)in_sizes; (void)n_in; (void)out_size;
  const float* x   = (const float*)d_in[0];   // [2048, 512]
  const float* W   = (const float*)d_in[1];   // [32768, 512]
  const int*   cat = (const int*)d_in[2];     // [2048]
  float* out = (float*)d_out;

  float*  ws     = (float*)d_ws;
  float*  tot    = ws;
  float*  rsinv  = ws + B_ROWS;
  float*  accums = ws + B_ROWS + CK;
  __bf16* xbf    = (__bf16*)(ws + XBF_OFF_F);
  __bf16* Wbf    = (__bf16*)(ws + WBF_OFF_F);

  const bool preconv = (ws_size >= WS_NEED_BYTES);  // fixed per harness

  zero_kernel<<<(B_ROWS + 255) / 256, 256, 0, stream>>>(tot, accums);
  wstats_kernel<<<CATS / 8, 256, 0, stream>>>(W, rsinv, accums);

  dim3 gTot(CK / 128, B_ROWS / 64), gLoss(B_ROWS / 32, B_ROWS / 64);
  if (preconv) {
    cvt_kernel<<<(B_ROWS * DIM / 8 + 255) / 256, 256, 0, stream>>>(x, xbf, B_ROWS * DIM);
    cvt_kernel<<<(CK * DIM / 8 + 255) / 256, 256, 0, stream>>>(W, Wbf, CK * DIM);
    gemm_kernel<true, false><<<gTot, 256, 0, stream>>>(xbf, Wbf, cat, rsinv, tot, accums);
    gemm_kernel<true, true><<<gLoss, 256, 0, stream>>>(xbf, Wbf, cat, rsinv, tot, accums);
  } else {
    gemm_kernel<false, false><<<gTot, 256, 0, stream>>>(x, W, cat, rsinv, tot, accums);
    gemm_kernel<false, true><<<gLoss, 256, 0, stream>>>(x, W, cat, rsinv, tot, accums);
  }
  finalize_kernel<<<1, 1, 0, stream>>>(accums, out);
}
